// Generator_51960514347232
// MI455X (gfx1250) — matrix operand order repaired and checked by reading, not compiled, not run on hardware
//
#include <hip/hip_runtime.h>
#include <hip/hip_bf16.h>

// ---------------------------------------------------------------------------
// Problem dims (from the reference)
// ---------------------------------------------------------------------------
#define B_DIM   512
#define T_DIM   256
#define L_DIM   256
#define HE_DIM  256
#define HT_DIM  512
#define OUT_DIM 88
#define OUTP    96   // OUT padded to 6 WMMA column tiles

typedef unsigned short u16t;
typedef unsigned int   u32t;
typedef __attribute__((ext_vector_type(16))) __bf16 v16bf;
typedef __attribute__((ext_vector_type(2)))  __bf16 v2bf;
typedef __attribute__((ext_vector_type(8)))  float  v8f;
typedef __attribute__((ext_vector_type(8)))  u32t   v8u;

// CDNA5 async global->LDS path (guarded: falls back to direct global loads)
#if defined(__HIP_DEVICE_COMPILE__) && \
    __has_builtin(__builtin_amdgcn_global_load_async_to_lds_b128) && \
    __has_builtin(__builtin_amdgcn_s_wait_asynccnt)
#define USE_ASYNC 1
typedef __attribute__((ext_vector_type(4))) int i32x4;
typedef __attribute__((address_space(1))) i32x4 g_i32x4;   // global
typedef __attribute__((address_space(3))) i32x4 l_i32x4;   // LDS
#else
#define USE_ASYNC 0
#endif

// ---------------------------------------------------------------------------
// bf16 helpers: prefer hardware packed convert; else compiler fptrunc (the
// backend knows the native f32->bf16 convert); never the manual bit-twiddle.
// ---------------------------------------------------------------------------
#if defined(__HIP_DEVICE_COMPILE__) && __has_builtin(__builtin_amdgcn_cvt_pk_bf16_f32)
__device__ __forceinline__ u32t pack2(float lo, float hi) {
  auto p = __builtin_amdgcn_cvt_pk_bf16_f32(lo, hi);   // v_cvt_pk_bf16_f32
  return __builtin_bit_cast(u32t, p);
}
__device__ __forceinline__ u16t f2bf(float x) {
  auto p = __builtin_amdgcn_cvt_pk_bf16_f32(x, 0.0f);
  return (u16t)__builtin_bit_cast(u32t, p);
}
#else
__device__ __forceinline__ u32t pack2(float lo, float hi) {
  v2bf p; p[0] = (__bf16)lo; p[1] = (__bf16)hi;
  return __builtin_bit_cast(u32t, p);
}
__device__ __forceinline__ u16t f2bf(float x) {
  __bf16 b = (__bf16)x;
  return __builtin_bit_cast(u16t, b);
}
#endif

__device__ __forceinline__ v8f splat8(float v) {
  v8f r;
#pragma unroll
  for (int i = 0; i < 8; ++i) r[i] = v;
  return r;
}
__device__ __forceinline__ float sigm(float x) { return 1.0f / (1.0f + __expf(-x)); }
__device__ __forceinline__ float softplusf(float x) {
  return x > 20.0f ? x : __logf(1.0f + __expf(x));
}

// Two 16B loads: elements [0..7] at p, [8..15] at p+16 (A-fragment K pattern)
__device__ __forceinline__ v8u load_split16(const u16t* p) {
  uint4 a = *(const uint4*)(p);
  uint4 b = *(const uint4*)(p + 16);
  v8u r;
  r[0] = a.x; r[1] = a.y; r[2] = a.z; r[3] = a.w;
  r[4] = b.x; r[5] = b.y; r[6] = b.z; r[7] = b.w;
  return r;
}

// A-matrix fragment (16x32 bf16) from an LDS tile, row stride in ushorts.
// ISA layout: lane ln holds row ln; K = k0 + 8*hi + {0..7} and {16..23}.
__device__ __forceinline__ v8u a_frag(const u16t* buf, int stride, int ln, int k0, int hi) {
  return load_split16(buf + ln * stride + k0 + hi * 8);
}

// A-matrix fragment straight from a row-major fp32 matrix (on-the-fly bf16).
__device__ __forceinline__ v8u a_frag_g(const float* row, int k0, int hi) {
  const float4* p = (const float4*)(row + k0 + hi * 8);
  float4 x0 = p[0], x1 = p[1];   // K = k0+8hi    .. +7
  float4 y0 = p[4], y1 = p[5];   // K = k0+8hi+16 .. +23
  v8u r;
  r[0] = pack2(x0.x, x0.y); r[1] = pack2(x0.z, x0.w);
  r[2] = pack2(x1.x, x1.y); r[3] = pack2(x1.z, x1.w);
  r[4] = pack2(y0.x, y0.y); r[5] = pack2(y0.z, y0.w);
  r[6] = pack2(y1.x, y1.y); r[7] = pack2(y1.z, y1.w);
  return r;
}

// B-matrix fragment (32x16 bf16) from a row-major bf16 weight [K][ldn].
// ISA layout: lane holds column N = n0 + ln; K = k0 + 8*hi + {0..7} and {16..23}.
__device__ __forceinline__ v8u b_frag(const u16t* W, int ldn, int k0, int n0, int lane) {
  int kb = k0 + ((lane >> 4) << 3);
  const u16t* p = W + (size_t)kb * ldn + n0 + (lane & 15);
  v8u r;
#pragma unroll
  for (int i = 0; i < 4; ++i)
    r[i] = (u32t)p[(size_t)(2 * i) * ldn] | ((u32t)p[(size_t)(2 * i + 1) * ldn] << 16);
#pragma unroll
  for (int i = 4; i < 8; ++i)
    r[i] = (u32t)p[(size_t)(8 + 2 * i) * ldn] | ((u32t)p[(size_t)(9 + 2 * i) * ldn] << 16);
  return r;
}

__device__ __forceinline__ v8f wmma_bf16(v8u a, v8u b, v8f c) {
  return __builtin_amdgcn_wmma_f32_16x16x32_bf16(
      /*neg_a=*/false, __builtin_bit_cast(v16bf, a),
      /*neg_b=*/false, __builtin_bit_cast(v16bf, b),
      /*c_mod=*/(short)0, c, /*reuse_a=*/false, /*reuse_b=*/false);
}

// K=256 GEMM pass: hoist all 8 B fragments (8 x 16 strided 16-bit loads),
// then run 8 back-to-back WMMAs against pre-hoisted A fragments.
__device__ __forceinline__ v8f gemm_k256(const v8u* af, const u16t* W, int ldn,
                                         int n0, int lane, v8f acc) {
  v8u bfr[8];
#pragma unroll
  for (int k = 0; k < 8; ++k) bfr[k] = b_frag(W, ldn, k * 32, n0, lane);
#pragma unroll
  for (int k = 0; k < 8; ++k) acc = wmma_bf16(af[k], bfr[k], acc);
  return acc;
}

// ---------------------------------------------------------------------------
// Weight conversion (fp32 -> bf16) into workspace
// ---------------------------------------------------------------------------
__global__ void cvt_kernel(const float* __restrict__ src, u16t* __restrict__ dst, int n) {
  int i = blockIdx.x * blockDim.x + threadIdx.x;
  if (i < n) dst[i] = f2bf(src[i]);
}
// We3: [256,88] -> zero-padded [256,96]
__global__ void cvt_pad_kernel(const float* __restrict__ src, u16t* __restrict__ dst) {
  int i = blockIdx.x * blockDim.x + threadIdx.x;
  if (i < HE_DIM * OUTP) {
    int k = i / OUTP, n = i - k * OUTP;
    dst[i] = (n < OUT_DIM) ? f2bf(src[k * OUT_DIM + n]) : (u16t)0;
  }
}

// ---------------------------------------------------------------------------
// Emission MLP: x_hat = sigmoid(relu(relu(z@W1+b1)@W2+b2)@W3+b3)
// 1024 blocks x 256 threads; each wave owns 16 rows of [B*T, L] and flows
// them through all 3 layers using a wave-private 16x256 bf16 LDS tile.
// ---------------------------------------------------------------------------
__global__ __launch_bounds__(256) void emission_kernel(
    const float* __restrict__ z_hat,
    const u16t* __restrict__ W1, const float* __restrict__ b1,
    const u16t* __restrict__ W2, const float* __restrict__ b2,
    const u16t* __restrict__ W3, const float* __restrict__ b3,
    float* __restrict__ x_hat) {
  __shared__ u16t Act[8 * 16 * L_DIM];   // 64 KB: one 16x256 bf16 tile per wave

  const int tid  = threadIdx.x;
  const int wave = tid >> 5, lane = tid & 31;
  const int hi = lane >> 4, ln = lane & 15;
  const size_t row_base = (size_t)blockIdx.x * 128 + (size_t)wave * 16;
  u16t* act = Act + wave * 16 * L_DIM;

  v8u af[8];

  // ---- layer 1: A from global fp32 (converted on the fly) ----------------
#pragma unroll
  for (int k = 0; k < 8; ++k)
    af[k] = a_frag_g(z_hat + (row_base + ln) * L_DIM, k * 32, hi);
  for (int nt = 0; nt < 16; ++nt) {
    int n0 = nt * 16;
    v8f acc = gemm_k256(af, W1, HE_DIM, n0, lane, splat8(b1[n0 + ln]));
#pragma unroll
    for (int r = 0; r < 8; ++r)
      act[(r + 8 * hi) * L_DIM + n0 + ln] = f2bf(fmaxf(acc[r], 0.0f));
  }

  // ---- layer 2: hoist A frags, then overwrite the same tile --------------
#pragma unroll
  for (int k = 0; k < 8; ++k) af[k] = a_frag(act, L_DIM, ln, k * 32, hi);
  for (int nt = 0; nt < 16; ++nt) {
    int n0 = nt * 16;
    v8f acc = gemm_k256(af, W2, HE_DIM, n0, lane, splat8(b2[n0 + ln]));
#pragma unroll
    for (int r = 0; r < 8; ++r)
      act[(r + 8 * hi) * L_DIM + n0 + ln] = f2bf(fmaxf(acc[r], 0.0f));
  }

  // ---- layer 3: 96 padded cols, sigmoid, masked store --------------------
#pragma unroll
  for (int k = 0; k < 8; ++k) af[k] = a_frag(act, L_DIM, ln, k * 32, hi);
  for (int nt = 0; nt < 6; ++nt) {
    int n0 = nt * 16, col = n0 + ln;
    v8f acc = gemm_k256(af, W3, OUTP, n0, lane,
                        splat8(col < OUT_DIM ? b3[col] : 0.0f));
    if (col < OUT_DIM) {
#pragma unroll
      for (int r = 0; r < 8; ++r)
        x_hat[(row_base + r + 8 * hi) * OUT_DIM + col] = sigm(acc[r]);
    }
  }
}

// ---------------------------------------------------------------------------
// Transition scan: 32 blocks, each owns 16 batch rows for all 256 timesteps.
// Carry z lives in LDS as bf16; weights stream from L2 (1.3 MB bf16 total).
// Next step's noise tile is prefetched with async global->LDS copies.
// ---------------------------------------------------------------------------
__global__ __launch_bounds__(256) void scan_kernel(
    const float* __restrict__ noise, const float* __restrict__ z_init,
    const u16t* __restrict__ Wg1, const float* __restrict__ bg1,
    const u16t* __restrict__ Wg2, const float* __restrict__ bg2,
    const u16t* __restrict__ Wh1, const float* __restrict__ bh1,
    const u16t* __restrict__ Wh2, const float* __restrict__ bh2,
    const u16t* __restrict__ Wmu, const float* __restrict__ bmu,
    const u16t* __restrict__ Ws,  const float* __restrict__ bs,
    float* __restrict__ mus, float* __restrict__ sigmas) {
  __shared__ u16t zbuf[16 * L_DIM];    // carry z_t              (8 KB)
  __shared__ u16t Ubuf[16 * HT_DIM];   // hidden (gate / H path) (16 KB)
  __shared__ u16t hbuf[16 * L_DIM];    // relu(hh)               (8 KB)
#if USE_ASYNC
  __shared__ float ebuf[2][16 * L_DIM];  // double-buffered noise tile (32 KB)
#endif

  const int tid = threadIdx.x;
  const int b0  = blockIdx.x * 16;
  const int wave = tid >> 5, lane = tid & 31;
  const int hi = lane >> 4, ln = lane & 15;

  // warm the weight stream into cache
  __builtin_prefetch(Wg1 + (size_t)tid * 64, 0, 1);
  __builtin_prefetch(Wh1 + (size_t)tid * 64, 0, 1);

#if USE_ASYNC
  // prefetch noise tile for t=0 (rows b0..b0+15 are contiguous: 16*256 floats)
  {
    const float* gsrc = noise + (size_t)b0 * L_DIM;
#pragma unroll
    for (int q = 0; q < 4; ++q) {
      int j = (tid + q * 256) * 4;
      __builtin_amdgcn_global_load_async_to_lds_b128(
          (g_i32x4*)(gsrc + j), (l_i32x4*)(&ebuf[0][j]), 0, 0);
    }
  }
#endif

  // stage z_init rows b0..b0+15 as bf16
  for (int i = tid; i < 16 * 64; i += 256) {
    int r = i >> 6, c = (i & 63) << 2;
    float4 f = *(const float4*)(z_init + (size_t)(b0 + r) * L_DIM + c);
    uint2 w; w.x = pack2(f.x, f.y); w.y = pack2(f.z, f.w);
    *(uint2*)(&zbuf[r * L_DIM + c]) = w;
  }
  __syncthreads();

  for (int t = 0; t < T_DIM; ++t) {
#if USE_ASYNC
    // kick off the async prefetch of the NEXT step's noise tile
    {
      int tn = (t + 1) & (T_DIM - 1);
      const float* gsrc = noise + (size_t)tn * (B_DIM * L_DIM) + (size_t)b0 * L_DIM;
      float* ldst = ebuf[(t + 1) & 1];
#pragma unroll
      for (int q = 0; q < 4; ++q) {
        int j = (tid + q * 256) * 4;
        __builtin_amdgcn_global_load_async_to_lds_b128(
            (g_i32x4*)(gsrc + j), (l_i32x4*)(ldst + j), 0, 0);
      }
    }
#endif

    // hoist the shared 16x256 z tile as A fragments (reused by 3 GEMMs)
    v8u za[8];
#pragma unroll
    for (int k = 0; k < 8; ++k) za[k] = a_frag(zbuf, L_DIM, ln, k * 32, hi);

    // ---- gate hidden: relu(z @ Wg1 + bg1) -> Ubuf [16,512] --------------
    for (int j = 0; j < 4; ++j) {
      int n0 = (wave + j * 8) * 16;
      v8f acc = gemm_k256(za, Wg1, HT_DIM, n0, lane, splat8(bg1[n0 + ln]));
#pragma unroll
      for (int r = 0; r < 8; ++r)
        Ubuf[(r + 8 * hi) * HT_DIM + n0 + ln] = f2bf(fmaxf(acc[r], 0.0f));
    }
    __syncthreads();

    // ---- g = sigmoid(U @ Wg2 + bg2): wave keeps its 2 col tiles in regs --
    v8f gv[2];
    for (int j = 0; j < 2; ++j) {
      int n0 = (2 * wave + j) * 16;
      v8f acc = splat8(bg2[n0 + ln]);
#pragma unroll
      for (int h = 0; h < 2; ++h) {     // K = 512 in two hoisted halves
        v8u bfr[8];
#pragma unroll
        for (int k = 0; k < 8; ++k) bfr[k] = b_frag(Wg2, L_DIM, (h * 8 + k) * 32, n0, lane);
#pragma unroll
        for (int k = 0; k < 8; ++k) {
          v8u a = a_frag(Ubuf, HT_DIM, ln, (h * 8 + k) * 32, hi);
          acc = wmma_bf16(a, bfr[k], acc);
        }
      }
#pragma unroll
      for (int r = 0; r < 8; ++r) acc[r] = sigm(acc[r]);
      gv[j] = acc;
    }
    __syncthreads();

    // ---- H hidden: relu(z @ Wh1 + bh1) -> Ubuf (reuse) -------------------
    for (int j = 0; j < 4; ++j) {
      int n0 = (wave + j * 8) * 16;
      v8f acc = gemm_k256(za, Wh1, HT_DIM, n0, lane, splat8(bh1[n0 + ln]));
#pragma unroll
      for (int r = 0; r < 8; ++r)
        Ubuf[(r + 8 * hi) * HT_DIM + n0 + ln] = f2bf(fmaxf(acc[r], 0.0f));
    }
    __syncthreads();

    // ---- hh = U2 @ Wh2 + bh2 ; zmu = z @ Wmu + bmu (both in regs) --------
    v8f hv[2], mz[2];
    for (int j = 0; j < 2; ++j) {
      int n0 = (2 * wave + j) * 16;
      v8f acc = splat8(bh2[n0 + ln]);
#pragma unroll
      for (int h = 0; h < 2; ++h) {     // K = 512 in two hoisted halves
        v8u bfr[8];
#pragma unroll
        for (int k = 0; k < 8; ++k) bfr[k] = b_frag(Wh2, L_DIM, (h * 8 + k) * 32, n0, lane);
#pragma unroll
        for (int k = 0; k < 8; ++k) {
          v8u a = a_frag(Ubuf, HT_DIM, ln, (h * 8 + k) * 32, hi);
          acc = wmma_bf16(a, bfr[k], acc);
        }
      }
      hv[j] = acc;
      mz[j] = gemm_k256(za, Wmu, L_DIM, n0, lane, splat8(bmu[n0 + ln]));
    }
    // stage relu(hh) for the sigma GEMM (needs all 256 cols of hh)
    for (int j = 0; j < 2; ++j) {
      int n0 = (2 * wave + j) * 16;
#pragma unroll
      for (int r = 0; r < 8; ++r)
        hbuf[(r + 8 * hi) * L_DIM + n0 + ln] = f2bf(fmaxf(hv[j][r], 0.0f));
    }
#if USE_ASYNC
    // Async loads complete in order: waiting down to 4 outstanding (the t+1
    // group) guarantees step-t's noise tile is resident in LDS. The barrier
    // below then publishes it workgroup-wide.
    __builtin_amdgcn_s_wait_asynccnt(4);
#endif
    __syncthreads();

    // ---- sigma = softplus(relu(hh) @ Ws + bs); outputs; z update ---------
    v8u ha[8];
#pragma unroll
    for (int k = 0; k < 8; ++k) ha[k] = a_frag(hbuf, L_DIM, ln, k * 32, hi);
    for (int j = 0; j < 2; ++j) {
      int n0 = (2 * wave + j) * 16, col = n0 + ln;
      v8f acc = gemm_k256(ha, Ws, L_DIM, n0, lane, splat8(bs[col]));
#pragma unroll
      for (int r = 0; r < 8; ++r) {
        int row = r + 8 * hi;
        int b   = b0 + row;
        float g = gv[j][r];
        float m = hv[j][r] * g + mz[j][r] * (1.0f - g);
        float s = softplusf(acc[r]);
        size_t o = (size_t)b * (T_DIM * L_DIM) + (size_t)t * L_DIM + col;
        mus[o]    = m;
        sigmas[o] = s;
#if USE_ASYNC
        float eps = ebuf[t & 1][row * L_DIM + col];
#else
        float eps = noise[(size_t)t * (B_DIM * L_DIM) + (size_t)b * L_DIM + col];
#endif
        zbuf[row * L_DIM + col] = f2bf(m + __fsqrt_rn(s) * eps);
      }
    }
    __syncthreads();
  }
}

// ---------------------------------------------------------------------------
// Launch
// ---------------------------------------------------------------------------
extern "C" void kernel_launch(void* const* d_in, const int* in_sizes, int n_in,
                              void* d_out, int out_size, void* d_ws, size_t ws_size,
                              hipStream_t stream) {
  const float* z_hat  = (const float*)d_in[0];
  const float* noise  = (const float*)d_in[1];
  const float* z_init = (const float*)d_in[2];
  const float* We1 = (const float*)d_in[3];  const float* be1 = (const float*)d_in[4];
  const float* We2 = (const float*)d_in[5];  const float* be2 = (const float*)d_in[6];
  const float* We3 = (const float*)d_in[7];  const float* be3 = (const float*)d_in[8];
  const float* Wg1 = (const float*)d_in[9];  const float* bg1 = (const float*)d_in[10];
  const float* Wg2 = (const float*)d_in[11]; const float* bg2 = (const float*)d_in[12];
  const float* Wh1 = (const float*)d_in[13]; const float* bh1 = (const float*)d_in[14];
  const float* Wh2 = (const float*)d_in[15]; const float* bh2 = (const float*)d_in[16];
  const float* Wmu = (const float*)d_in[17]; const float* bmu = (const float*)d_in[18];
  const float* Wsg = (const float*)d_in[19]; const float* bsg = (const float*)d_in[20];

  // bf16 weight cache layout in the workspace (element offsets)
  u16t* ws = (u16t*)d_ws;
  u16t* Wg1b = ws + 0;        // 256*512
  u16t* Wg2b = ws + 131072;   // 512*256
  u16t* Wh1b = ws + 262144;   // 256*512
  u16t* Wh2b = ws + 393216;   // 512*256
  u16t* Wmub = ws + 524288;   // 256*256
  u16t* Wsb  = ws + 589824;   // 256*256
  u16t* We1b = ws + 655360;   // 256*256
  u16t* We2b = ws + 720896;   // 256*256
  u16t* We3b = ws + 786432;   // 256*96 (padded)

  auto cvt = [&](const float* s, u16t* d, int n) {
    cvt_kernel<<<(n + 255) / 256, 256, 0, stream>>>(s, d, n);
  };
  cvt(Wg1, Wg1b, L_DIM * HT_DIM);
  cvt(Wg2, Wg2b, HT_DIM * L_DIM);
  cvt(Wh1, Wh1b, L_DIM * HT_DIM);
  cvt(Wh2, Wh2b, HT_DIM * L_DIM);
  cvt(Wmu, Wmub, L_DIM * L_DIM);
  cvt(Wsg, Wsb,  L_DIM * L_DIM);
  cvt(We1, We1b, L_DIM * HE_DIM);
  cvt(We2, We2b, HE_DIM * HE_DIM);
  cvt_pad_kernel<<<(HE_DIM * OUTP + 255) / 256, 256, 0, stream>>>(We3, We3b);

  float* x_hat  = (float*)d_out;
  float* mus    = x_hat + (size_t)B_DIM * T_DIM * OUT_DIM;
  float* sigmas = mus   + (size_t)B_DIM * T_DIM * L_DIM;

  emission_kernel<<<(B_DIM * T_DIM) / 128, 256, 0, stream>>>(
      z_hat, We1b, be1, We2b, be2, We3b, be3, x_hat);
  scan_kernel<<<B_DIM / 16, 256, 0, stream>>>(
      noise, z_init, Wg1b, bg1, Wg2b, bg2, Wh1b, bh1, Wh2b, bh2,
      Wmub, bmu, Wsb, bsg, mus, sigmas);
}
